// GGNN_61083024883836
// MI455X (gfx1250) — matrix operand order repaired and checked by reading, not compiled, not run on hardware
//
#include <hip/hip_runtime.h>
#include <hip/hip_bf16.h>

// GGNN: B=2, N=4096, E=2, T=4, D=32, ANN=32, steps=5, OUT=31
// Dominant op: a_in = A[b, 4096, 8192] x insT[b, 8192, 128]  (bf16 WMMA, f32 acc)
// B panel staged through LDS with CDNA5 async global->LDS DMA (ASYNCcnt).

#define Bx 2
#define Nx 4096
#define Ex 2
#define Tx 4
#define Dx 32
#define ANNx 32
#define Kx (Ex * Nx)      // 8192
#define COLS (Tx * Dx)    // 128
#define OUTx 31

typedef __attribute__((ext_vector_type(16))) __bf16 v16bf;
typedef __attribute__((ext_vector_type(8)))  float  v8f;
typedef __attribute__((ext_vector_type(4)))  int    v4i;

typedef __attribute__((address_space(1))) v4i* glb_v4i_p;
typedef __attribute__((address_space(3))) v4i* lds_v4i_p;

#if __has_builtin(__builtin_amdgcn_global_load_async_to_lds_b128)
#define ASYNC_LDS 1
#endif

#if __has_builtin(__builtin_amdgcn_s_wait_asynccnt)
#define WAIT_ASYNC(n) __builtin_amdgcn_s_wait_asynccnt(n)
#elif defined(ASYNC_LDS)
#define WAIT_ASYNC(n) asm volatile("s_wait_asynccnt %0" ::"i"(n) : "memory")
#else
#define WAIT_ASYNC(n)
#endif

__device__ __forceinline__ void copy16_to_lds(const void* g, void* l) {
#ifdef ASYNC_LDS
    __builtin_amdgcn_global_load_async_to_lds_b128((glb_v4i_p)g, (lds_v4i_p)l, 0, 0);
#else
    *(float4*)l = *(const float4*)g;
#endif
}

__device__ __forceinline__ float sigmoidf(float x) {
    return 1.0f / (1.0f + __expf(-x));
}

// ---------------------------------------------------------------------------
// Kernel 1: per-edge-type input linear.
//   ins[b,e,n,t,o] = b_in[e,o] + sum_d h[b,n,t,d] * W_in[e,o,d]
// written transposed + bf16:  insT[b][t*32+o][e*4096+n]
// ---------------------------------------------------------------------------
__global__ void ggnn_ins_kernel(const float* __restrict__ h,
                                const float* __restrict__ W_in,
                                const float* __restrict__ b_in,
                                __bf16* __restrict__ insT) {
    int idx = blockIdx.x * 256 + threadIdx.x;      // B*E*N*T*D = 2,097,152
    int o = idx & 31;
    int t = (idx >> 5) & 3;
    int n = (idx >> 7) & (Nx - 1);
    int e = (idx >> 19) & 1;
    int b = idx >> 20;

    const float* hp = h + ((size_t)((b * Nx + n) * Tx + t)) * Dx;
    const float* wp = W_in + (size_t)(e * Dx + o) * Dx;
    float acc = b_in[e * Dx + o];
#pragma unroll
    for (int d = 0; d < Dx; ++d) acc += hp[d] * wp[d];

    insT[((size_t)b * COLS + (t * Dx + o)) * Kx + (e * Nx + n)] = (__bf16)acc;
}

// ---------------------------------------------------------------------------
// Kernel 2: WMMA bf16 GEMM  a_in[b][n][col] = sum_m A[b][n][m] * insT[b][col][m]
// Block = 8 waves, 8 consecutive 16-row tiles of one batch; all waves share
// the B panel, which is double-buffer staged in LDS via async global->LDS.
// A is converted f32 -> bf16 in registers.
// ---------------------------------------------------------------------------
__global__ void ggnn_aggr_wmma_kernel(const float* __restrict__ A,
                                      const __bf16* __restrict__ insT,
                                      float* __restrict__ a_in) {
    // one k-chunk of B: 128 cols x 32 K x bf16 = 8 KB, layout [col][k] (64B/col)
    __shared__ alignas(32) unsigned char ldsB[2][COLS * 64];

    int tid  = threadIdx.x;
    int wave = tid >> 5;
    int lane = tid & 31;
    int lh   = lane & 15;        // M row / N col within tile
    int hi   = lane >> 4;        // K-half selector

    int b     = blockIdx.x >> 5;                  // 32 blocks per batch (uniform)
    int tile  = blockIdx.x * 8 + wave;            // 0..511
    int mTile = tile & 255;

    const float*  Arow = A + (size_t)(b * Nx + mTile * 16 + lh) * Kx;
    const __bf16* insB = insT + (size_t)b * COLS * Kx;

    v8f acc[8] = {};

    // stage chunk 0: 512 x b128 transfers, 256 threads x 2 issues
#pragma unroll
    for (int it = 0; it < 2; ++it) {
        int j   = it * 256 + tid;
        int col = j >> 2;
        int seg = j & 3;
        copy16_to_lds(insB + (size_t)col * Kx + seg * 8,
                      &ldsB[0][col * 64 + seg * 16]);
    }

    for (int kc = 0; kc < 256; ++kc) {
        int cur = kc & 1;

        // ---- A fragment loads (global f32), overlap with async wait ----
        int ka = kc * 32 + hi * 16;
        const float4* Ap = reinterpret_cast<const float4*>(Arow + ka);
        float4 a0 = Ap[0];   // K = ka .. ka+3
        float4 a1 = Ap[1];   // K = ka+4 .. ka+7
        float4 a2 = Ap[2];   // K = ka+8 .. ka+11
        float4 a3 = Ap[3];   // K = ka+12 .. ka+15

        // ---- prefetch next B chunk, retire current one ----
        if (kc < 255) {
#pragma unroll
            for (int it = 0; it < 2; ++it) {
                int j   = it * 256 + tid;
                int col = j >> 2;
                int seg = j & 3;
                copy16_to_lds(insB + (size_t)col * Kx + (kc + 1) * 32 + seg * 8,
                              &ldsB[cur ^ 1][col * 64 + seg * 16]);
            }
            WAIT_ASYNC(2);   // in-order: chunk kc complete, kc+1 in flight
        } else {
            WAIT_ASYNC(0);
        }

        v16bf af;
        af[0]  = (__bf16)a0.x; af[1]  = (__bf16)a0.y;
        af[2]  = (__bf16)a0.z; af[3]  = (__bf16)a0.w;
        af[4]  = (__bf16)a1.x; af[5]  = (__bf16)a1.y;
        af[6]  = (__bf16)a1.z; af[7]  = (__bf16)a1.w;
        af[8]  = (__bf16)a2.x; af[9]  = (__bf16)a2.y;
        af[10] = (__bf16)a2.z; af[11] = (__bf16)a2.w;
        af[12] = (__bf16)a3.x; af[13] = (__bf16)a3.y;
        af[14] = (__bf16)a3.z; af[15] = (__bf16)a3.w;

        __syncthreads();     // all waves' async data for chunk kc visible

        // ---- preload all 8 B fragments into distinct registers so the
        //      ds_load_b128s form one clause and pipeline ahead of the WMMAs
        v16bf bf[8];
#pragma unroll
        for (int c = 0; c < 8; ++c) {
            bf[c] = *reinterpret_cast<const v16bf*>(
                &ldsB[cur][(c * 16 + lh) * 64 + hi * 32]);
        }
#pragma unroll
        for (int c = 0; c < 8; ++c) {
            acc[c] = __builtin_amdgcn_wmma_f32_16x16x32_bf16(
                false, af, false, bf[c], (short)0, acc[c], false, false);
        }

        __syncthreads();     // buffer reuse safe before next prefetch lands
    }

    // ---- store: acc[c][r] -> row M = hi*8 + r, col N = lh ----
    float* Crow = a_in + (size_t)(b * Nx + mTile * 16) * COLS;
#pragma unroll
    for (int c = 0; c < 8; ++c) {
#pragma unroll
        for (int r = 0; r < 8; ++r) {
            Crow[(size_t)(hi * 8 + r) * COLS + c * 16 + lh] = acc[c][r];
        }
    }
}

// ---------------------------------------------------------------------------
// Kernel 3: GRU-style update. One wave per (b,n,t) row, lane = d.
// ---------------------------------------------------------------------------
__global__ void ggnn_gru_kernel(const float* __restrict__ a_in,
                                const float* __restrict__ h,
                                const float* __restrict__ Wr, const float* __restrict__ br,
                                const float* __restrict__ Wz, const float* __restrict__ bz,
                                const float* __restrict__ Wh, const float* __restrict__ bh,
                                float* __restrict__ h_next) {
    __shared__ float xs[8][2 * Dx];
    __shared__ float rhs[8][Dx];

    int w = threadIdx.x >> 5;
    int d = threadIdx.x & 31;
    int row = blockIdx.x * 8 + w;            // B*N*T = 32768 rows
    size_t base = (size_t)row * Dx;

    xs[w][d]      = a_in[base + d];
    xs[w][Dx + d] = h[base + d];
    __syncthreads();

    float accr = br[d], accz = bz[d];
#pragma unroll 8
    for (int k = 0; k < 2 * Dx; ++k) {
        float x = xs[w][k];
        accr += Wr[d * 2 * Dx + k] * x;
        accz += Wz[d * 2 * Dx + k] * x;
    }
    float r = sigmoidf(accr);
    float z = sigmoidf(accz);
    rhs[w][d] = r * xs[w][Dx + d];
    __syncthreads();

    float acch = bh[d];
#pragma unroll 8
    for (int k = 0; k < Dx; ++k) acch += Wh[d * 2 * Dx + k] * xs[w][k];
#pragma unroll 8
    for (int k = 0; k < Dx; ++k) acch += Wh[d * 2 * Dx + Dx + k] * rhs[w][k];
    float hh = tanhf(acch);

    h_next[base + d] = (1.0f - z) * xs[w][Dx + d] + z * hh;
}

// ---------------------------------------------------------------------------
// Kernel 4: output head. One wave per (b,n).
// ---------------------------------------------------------------------------
__global__ void ggnn_out_kernel(const float* __restrict__ h,
                                const float* __restrict__ ann,
                                const float* __restrict__ Wj, const float* __restrict__ bj,
                                const float* __restrict__ Wout, const float* __restrict__ bout,
                                float* __restrict__ out) {
    __shared__ float xs[8][Dx + ANNx];
    __shared__ float tmp[8][COLS];

    int w = threadIdx.x >> 5;
    int d = threadIdx.x & 31;
    int row = blockIdx.x * 8 + w;            // B*N = 8192 rows
    size_t base = (size_t)row * Tx * Dx;

    for (int t = 0; t < Tx; ++t) {
        __syncthreads();
        xs[w][d]      = h[base + (size_t)t * Dx + d];
        xs[w][Dx + d] = ann[base + (size_t)t * ANNx + d];
        __syncthreads();
        float acc = bj[d];
#pragma unroll 8
        for (int k = 0; k < Dx + ANNx; ++k) acc += Wj[d * (Dx + ANNx) + k] * xs[w][k];
        tmp[w][t * Dx + d] = tanhf(acc);
    }
    __syncthreads();

    if (d < OUTx) {
        float acc = bout[d];
#pragma unroll 8
        for (int k = 0; k < COLS; ++k) acc += Wout[d * COLS + k] * tmp[w][k];
        out[(size_t)row * OUTx + d] = acc;
    }
}

// ---------------------------------------------------------------------------
extern "C" void kernel_launch(void* const* d_in, const int* in_sizes, int n_in,
                              void* d_out, int out_size, void* d_ws, size_t ws_size,
                              hipStream_t stream) {
    const float* prop = (const float*)d_in[0];
    const float* ann  = (const float*)d_in[1];
    const float* A    = (const float*)d_in[2];
    const float* W_in = (const float*)d_in[3];
    const float* b_in = (const float*)d_in[4];
    const float* Wr   = (const float*)d_in[5];
    const float* br   = (const float*)d_in[6];
    const float* Wz   = (const float*)d_in[7];
    const float* bz   = (const float*)d_in[8];
    const float* Wh   = (const float*)d_in[9];
    const float* bh   = (const float*)d_in[10];
    const float* Wj   = (const float*)d_in[11];
    const float* bj   = (const float*)d_in[12];
    const float* Wout = (const float*)d_in[13];
    const float* bout = (const float*)d_in[14];
    float* out = (float*)d_out;

    // workspace layout
    const size_t INS_BYTES = (size_t)Bx * COLS * Kx * sizeof(__bf16);   // 4 MB
    const size_t AIN_BYTES = (size_t)Bx * Nx * COLS * sizeof(float);    // 4 MB
    const size_t HN        = (size_t)Bx * Nx * Tx * Dx;                 // elems
    char* ws = (char*)d_ws;
    __bf16* insT = (__bf16*)ws;
    float*  a_in = (float*)(ws + INS_BYTES);
    float*  h0   = (float*)(ws + INS_BYTES + AIN_BYTES);
    float*  h1   = h0 + HN;

    const float* hcur = prop;
    float* hnext = h0;

    for (int step = 0; step < 5; ++step) {
        ggnn_ins_kernel<<<8192, 256, 0, stream>>>(hcur, W_in, b_in, insT);
        ggnn_aggr_wmma_kernel<<<64, 256, 0, stream>>>(A, insT, a_in);
        ggnn_gru_kernel<<<4096, 256, 0, stream>>>(a_in, hcur, Wr, br, Wz, bz, Wh, bh,
                                                  hnext);
        hcur = hnext;
        hnext = (hnext == h0) ? h1 : h0;
    }

    ggnn_out_kernel<<<1024, 256, 0, stream>>>(hcur, ann, Wj, bj, Wout, bout, out);
}
